// Griffin_41068477285012
// MI455X (gfx1250) — compile-verified, matrix operand order checked by reading, not run on hardware
//
#include <hip/hip_runtime.h>
#include <hip/hip_bf16.h>

typedef _Float16 v16h __attribute__((ext_vector_type(16)));
typedef _Float16 v8h  __attribute__((ext_vector_type(8)));
typedef _Float16 v4h  __attribute__((ext_vector_type(4)));
typedef float    v8f  __attribute__((ext_vector_type(8)));
typedef float    v4f  __attribute__((ext_vector_type(4)));

// gcc-style 16B int vector, matching the async-LDS builtin's parameter type
typedef int v4i_g __attribute__((vector_size(16)));
typedef __attribute__((address_space(1))) v4i_g* as1_v4i_p;
typedef __attribute__((address_space(3))) v4i_g* as3_v4i_p;

constexpr int Dm   = 1024;
constexpr int Tt   = 2048;
constexpr int Bb   = 4;
constexpr int NDEP = 2;
constexpr int WIN  = 128;
constexpr int Hh   = 4;
constexpr int HD   = 256;     // Dm / Hh
constexpr int BT   = Bb * Tt; // 8192 rows

// ---------------- epilogue modes ----------------
constexpr int EPI_F32    = 0;  // out_f32 = acc + bias
constexpr int EPI_SIG    = 1;  // out_f32 = sigmoid(acc + bias)
constexpr int EPI_GELU16 = 2;  // out_f16 = gelu(acc + bias)
constexpr int EPI_RES    = 3;  // out_f32 += acc + bias   (residual into x)

// ---------------- CDNA5 async global->LDS path ----------------
#if defined(__gfx1250__) && __has_builtin(__builtin_amdgcn_global_load_async_to_lds_b128)
#define USE_ASYNC_LDS 1
#else
#define USE_ASYNC_LDS 0
#endif

__device__ __forceinline__ void async_b128(const void* g, void* l)
{
#if USE_ASYNC_LDS
    // LDS generic address: low 32 bits are the LDS offset (aperture rule)
    __builtin_amdgcn_global_load_async_to_lds_b128(
        (as1_v4i_p)(unsigned long long)(size_t)g,
        (as3_v4i_p)(unsigned)(size_t)l, 0, 0);
#else
    (void)g; (void)l;
#endif
}

__device__ __forceinline__ void wait_async0()
{
#if defined(__gfx1250__)
#if __has_builtin(__builtin_amdgcn_s_wait_asynccnt)
    __builtin_amdgcn_s_wait_asynccnt(0);
#else
    asm volatile("s_wait_asynccnt 0x0" ::: "memory");
#endif
#endif
}

// =====================================================================
// WMMA GEMM:  C[M,N] = A[M,K] (f16, row-major) * Bt[N,K]^T (f16) + bias
// Block tile 64(M) x 256(N), 8 wave32s, each wave a 32x64 tile
// (2x4 v_wmma_f32_16x16x32_f16 -> 8 WMMA per K-step, A frags reused 4x).
// K stepped by 32 through double-buffered padded LDS; tiles staged with
// GLOBAL_LOAD_ASYNC_TO_LDS_B128 (ASYNCcnt) when available.
// M (8192), N (1024/3072/4096), K (1024/4096) all divide the tiles ->
// no divergence, EXEC all-ones for every WMMA.
// =====================================================================
template<int MODE>
__global__ __launch_bounds__(256)
void gemm_wmma(const _Float16* __restrict__ A, const _Float16* __restrict__ Bt,
               const float* __restrict__ bias, void* __restrict__ out,
               int M, int N, int K)
{
    constexpr int LDT = 40;              // LDS row stride in halves (80B, 16B-aligned)
    __shared__ _Float16 lA[2][64  * LDT];
    __shared__ _Float16 lB[2][256 * LDT];

    const int tid  = threadIdx.x;
    const int lane = tid & 31;
    const int wave = tid >> 5;           // 0..7
    const int li   = lane & 15;          // index within 16
    const int lh   = lane >> 4;          // half-select (K / row split)
    const int wm   = wave & 1;           // 0..1 -> 32-row slab
    const int wn   = wave >> 1;          // 0..3 -> 64-col slab

    const int bM = blockIdx.y * 64;
    const int bN = blockIdx.x * 256;

    // cooperative tile loaders: 256 thr, 16B (8 halves) per transfer
    const int  trow = tid >> 2;                  // 0..63
    const int  tchk = (tid & 3) * 8;             // 0/8/16/24 halves
    const long aG   = (long)(bM + trow) * K + tchk;
    long bG[4];
#pragma unroll
    for (int j = 0; j < 4; ++j)
        bG[j] = (long)(bN + trow + 64 * j) * K + tchk;

    const v8f zero = {0.f,0.f,0.f,0.f,0.f,0.f,0.f,0.f};
    v8f acc[2][4];
#pragma unroll
    for (int tm = 0; tm < 2; ++tm)
#pragma unroll
        for (int tn = 0; tn < 4; ++tn) acc[tm][tn] = zero;

    int cur = 0;

    // ---- stage tile k0=0 into buffer 0 ----
#if USE_ASYNC_LDS
    async_b128(A + aG, &lA[0][trow * LDT + tchk]);
#pragma unroll
    for (int j = 0; j < 4; ++j)
        async_b128(Bt + bG[j], &lB[0][(trow + 64 * j) * LDT + tchk]);
    wait_async0();
#else
    {
        v8h va = *(const v8h*)(A + aG);
        v8h vb[4];
#pragma unroll
        for (int j = 0; j < 4; ++j) vb[j] = *(const v8h*)(Bt + bG[j]);
        *(v8h*)&lA[0][trow * LDT + tchk] = va;
#pragma unroll
        for (int j = 0; j < 4; ++j)
            *(v8h*)&lB[0][(trow + 64 * j) * LDT + tchk] = vb[j];
    }
#endif
    __syncthreads();

    for (int k0 = 0; k0 < K; k0 += 32) {
        const bool has_next = (k0 + 32 < K);
        const int  nxt = cur ^ 1;

#if USE_ASYNC_LDS
        if (has_next) {                  // issue async stage of next K-slab
            async_b128(A + aG + k0 + 32, &lA[nxt][trow * LDT + tchk]);
#pragma unroll
            for (int j = 0; j < 4; ++j)
                async_b128(Bt + bG[j] + k0 + 32,
                           &lB[nxt][(trow + 64 * j) * LDT + tchk]);
        }
#else
        v8h va, vb[4];
        if (has_next) {                  // issue global loads of next K-slab
            va = *(const v8h*)(A + aG + k0 + 32);
#pragma unroll
            for (int j = 0; j < 4; ++j)
                vb[j] = *(const v8h*)(Bt + bG[j] + k0 + 32);
            __builtin_prefetch(A + aG + k0 + 64, 0, 1);
        }
#endif

        // ---- compute from lds[cur] while next slab is in flight ----
        // fragments: lanes 0-15 hold K {0..7,16..23}; lanes 16-31 {8..15,24..31}
        v16h af[2];
#pragma unroll
        for (int tm = 0; tm < 2; ++tm) {
            const int r = wm * 32 + tm * 16 + li;
            ((v8h*)&af[tm])[0] = *(const v8h*)&lA[cur][r * LDT + lh * 8];
            ((v8h*)&af[tm])[1] = *(const v8h*)&lA[cur][r * LDT + 16 + lh * 8];
        }
#pragma unroll
        for (int tn = 0; tn < 4; ++tn) {
            const int c = wn * 64 + tn * 16 + li;
            v16h bf;
            ((v8h*)&bf)[0] = *(const v8h*)&lB[cur][c * LDT + lh * 8];
            ((v8h*)&bf)[1] = *(const v8h*)&lB[cur][c * LDT + 16 + lh * 8];
#pragma unroll
            for (int tm = 0; tm < 2; ++tm)
                acc[tm][tn] = __builtin_amdgcn_wmma_f32_16x16x32_f16(
                    false, af[tm], false, bf, (short)0, acc[tm][tn],
                    false, false);
        }

#if USE_ASYNC_LDS
        wait_async0();                   // my async writes into lds[nxt] done
        __syncthreads();                 // everyone done reading lds[cur] too
#else
        if (has_next) {
            *(v8h*)&lA[nxt][trow * LDT + tchk] = va;
#pragma unroll
            for (int j = 0; j < 4; ++j)
                *(v8h*)&lB[nxt][(trow + 64 * j) * LDT + tchk] = vb[j];
        }
        __syncthreads();
#endif
        cur = nxt;
    }

    // epilogue: C/D layout -> row = base + vgpr + 8*lh, col = base + lane%16
#pragma unroll
    for (int tm = 0; tm < 2; ++tm) {
#pragma unroll
        for (int tn = 0; tn < 4; ++tn) {
            const int col = bN + wn * 64 + tn * 16 + li;
            const float bv = bias[col];
#pragma unroll
            for (int r = 0; r < 8; ++r) {
                const int  row = bM + wm * 32 + tm * 16 + lh * 8 + r;
                const long o   = (long)row * N + col;
                float v = acc[tm][tn][r] + bv;
                if (MODE == EPI_F32) {
                    ((float*)out)[o] = v;
                } else if (MODE == EPI_SIG) {
                    ((float*)out)[o] = 1.0f / (1.0f + __expf(-v));
                } else if (MODE == EPI_GELU16) {
                    ((_Float16*)out)[o] =
                        (_Float16)(0.5f * v * (1.0f + erff(v * 0.70710678118f)));
                } else { // EPI_RES
                    ((float*)out)[o] += v;
                }
            }
        }
    }
}

// =====================================================================
// LayerNorm over last dim (1024) -> f16 output (GEMM A operand)
// one 256-thread block per row, 4 elements/thread, LDS tree reduction
// =====================================================================
__global__ __launch_bounds__(256)
void ln_kernel(const float* __restrict__ x, const float* __restrict__ s,
               const float* __restrict__ b, _Float16* __restrict__ y)
{
    __shared__ float red[256];
    const int row = blockIdx.x;
    const int tid = threadIdx.x;
    const v4f v = *(const v4f*)(x + (long)row * Dm + tid * 4);

    float p = v[0] + v[1] + v[2] + v[3];
    red[tid] = p; __syncthreads();
    for (int o = 128; o > 0; o >>= 1) {
        if (tid < o) red[tid] += red[tid + o];
        __syncthreads();
    }
    const float mean = red[0] * (1.0f / Dm);
    __syncthreads();

    const float d0 = v[0]-mean, d1 = v[1]-mean, d2 = v[2]-mean, d3 = v[3]-mean;
    red[tid] = d0*d0 + d1*d1 + d2*d2 + d3*d3; __syncthreads();
    for (int o = 128; o > 0; o >>= 1) {
        if (tid < o) red[tid] += red[tid + o];
        __syncthreads();
    }
    const float rstd = rsqrtf(red[0] * (1.0f / Dm) + 1e-5f);

    v4h o16;
#pragma unroll
    for (int c = 0; c < 4; ++c) {
        const int col = tid * 4 + c;
        o16[c] = (_Float16)((v[c] - mean) * rstd * s[col] + b[col]);
    }
    *(v4h*)(y + (long)row * Dm + tid * 4) = o16;
}

// =====================================================================
// RG-LRU scan: h_t = g_t*h_{t-1} + (1-g_t)*u_t  (serial in T by nature)
// one lane per (b,d) channel; lanes are coalesced across d
// =====================================================================
__global__ __launch_bounds__(256)
void lru_scan(const float* __restrict__ g, const float* __restrict__ u,
              _Float16* __restrict__ h16)
{
    const int id = blockIdx.x * 256 + threadIdx.x;   // 0 .. B*D-1
    const int b  = id / Dm, d = id % Dm;
    float h = 0.f;
    for (int t = 0; t < Tt; ++t) {
        const long idx = ((long)(b * Tt + t)) * Dm + d;
        const float gv = g[idx];
        h = gv * h + (1.0f - gv) * u[idx];
        h16[idx] = (_Float16)h;
    }
}

// =====================================================================
// Windowed causal attention (WIN=128), flash-style online softmax.
// One wave per query (b,h,i); HD=256 -> 8 f32/lane; wave shfl reduction.
// Writes y as f16 (A operand of the attn_out GEMM).
// =====================================================================
__global__ __launch_bounds__(256)
void attn_kernel(const float* __restrict__ qkv, _Float16* __restrict__ y16)
{
    const int wid  = (blockIdx.x * 256 + threadIdx.x) >> 5;
    const int lane = threadIdx.x & 31;
    const int i = wid % Tt;
    const int h = (wid / Tt) % Hh;
    const int b = wid / (Tt * Hh);
    const float scale = 0.0625f;                     // 1/sqrt(256)

    const long rowQ = ((long)(b * Tt + i)) * (3 * Dm);
    float q[8];
#pragma unroll
    for (int e = 0; e < 8; ++e) q[e] = qkv[rowQ + h * HD + lane * 8 + e];

    float m = -INFINITY, wsum = 0.f;
    float accv[8] = {0.f,0.f,0.f,0.f,0.f,0.f,0.f,0.f};
    int j0 = i - (WIN - 1); if (j0 < 0) j0 = 0;

    for (int j = j0; j <= i; ++j) {
        const long rowK = ((long)(b * Tt + j)) * (3 * Dm);
        const float* kp = qkv + rowK + Dm     + h * HD + lane * 8;
        const float* vp = qkv + rowK + 2 * Dm + h * HD + lane * 8;
        float p = 0.f;
#pragma unroll
        for (int e = 0; e < 8; ++e) p += q[e] * kp[e];
#pragma unroll
        for (int off = 16; off >= 1; off >>= 1) p += __shfl_xor(p, off, 32);
        const float sc    = p * scale;
        const float mnew  = fmaxf(m, sc);
        const float alpha = __expf(m - mnew);        // first iter: exp(-inf)=0
        const float w     = __expf(sc - mnew);
#pragma unroll
        for (int e = 0; e < 8; ++e) accv[e] = accv[e] * alpha + w * vp[e];
        wsum = wsum * alpha + w;
        m = mnew;
    }
    const float inv = 1.0f / wsum;
    v8h o;
#pragma unroll
    for (int e = 0; e < 8; ++e) o[e] = (_Float16)(accv[e] * inv);
    *(v8h*)(y16 + ((long)(b * Tt + i)) * Dm + h * HD + lane * 8) = o;
}

// =====================================================================
// fp32 [K,N] weight  ->  f16 [N,K] transposed copy (one-time; result is
// 63 MB total, L2-resident on MI455X's 192 MB L2 for all later GEMMs)
// =====================================================================
__global__ __launch_bounds__(256)
void wconv_kernel(const float* __restrict__ w, _Float16* __restrict__ wT,
                  int K, int N)
{
    const long id = (long)blockIdx.x * 256 + threadIdx.x;   // over N*K
    if (id >= (long)K * N) return;
    const int n = (int)(id / K);
    const int k = (int)(id % K);
    wT[id] = (_Float16)w[(long)k * N + n];
}

// =====================================================================
extern "C" void kernel_launch(void* const* d_in, const int* in_sizes, int n_in,
                              void* d_out, int out_size, void* d_ws, size_t ws_size,
                              hipStream_t stream)
{
    (void)in_sizes; (void)n_in; (void)out_size; (void)ws_size;

    const float* x_in    = (const float*)d_in[0];
    const float* ln1_s   = (const float*)d_in[1];
    const float* ln1_b   = (const float*)d_in[2];
    const float* ln2_s   = (const float*)d_in[3];
    const float* ln2_b   = (const float*)d_in[4];
    const float* ln3_s   = (const float*)d_in[5];
    const float* ln3_b   = (const float*)d_in[6];
    const float* rg_in_w   = (const float*)d_in[7];
    const float* rg_in_b   = (const float*)d_in[8];
    const float* rg_gate_w = (const float*)d_in[9];
    const float* rg_gate_b = (const float*)d_in[10];
    const float* rg_out_w  = (const float*)d_in[11];
    const float* rg_out_b  = (const float*)d_in[12];
    const float* qkv_w     = (const float*)d_in[13];
    const float* qkv_b     = (const float*)d_in[14];
    const float* ao_w      = (const float*)d_in[15];
    const float* ao_b      = (const float*)d_in[16];
    const float* mlp_w1    = (const float*)d_in[17];
    const float* mlp_b1    = (const float*)d_in[18];
    const float* mlp_w2    = (const float*)d_in[19];
    const float* mlp_b2    = (const float*)d_in[20];

    // ---- workspace carve-up (256B aligned) ----
    char*  ws  = (char*)d_ws;
    size_t off = 0;
    auto take = [&](size_t bytes) -> char* {
        char* p = ws + off;
        off += (bytes + 255) & ~(size_t)255;
        return p;
    };
    float*     xbuf = (float*)    take((size_t)BT * Dm * 4);   // fp32 residual stream
    _Float16*  xln  = (_Float16*) take((size_t)BT * Dm * 2);   // f16 GEMM A operand
    float*     ubuf = (float*)    take((size_t)BT * Dm * 4);
    float*     gbuf = (float*)    take((size_t)BT * Dm * 4);
    char*      big  =             take((size_t)BT * 3 * Dm * 4); // qkv(f32) / mlp-hid(f16)
    float*     qkvb = (float*)big;
    _Float16*  hid  = (_Float16*)big;

    // transposed f16 weights: [rg_in, rg_gate, rg_out, qkv, attn_out, w1, w2]
    const size_t welem[7] = {(size_t)Dm*Dm, (size_t)Dm*Dm, (size_t)Dm*Dm,
                             (size_t)Dm*3*Dm, (size_t)Dm*Dm,
                             (size_t)Dm*4*Dm, (size_t)4*Dm*Dm};
    _Float16* wT[NDEP][7];
    for (int l = 0; l < NDEP; ++l)
        for (int m = 0; m < 7; ++m)
            wT[l][m] = (_Float16*)take(welem[m] * 2);

    // ---- one-time: x -> fp32 residual buffer; weights -> f16 transposed ----
    (void)hipMemcpyAsync(xbuf, x_in, (size_t)BT * Dm * 4, hipMemcpyDeviceToDevice, stream);

    auto wconv = [&](const float* w, _Float16* dst, int K, int N) {
        const long n = (long)K * N;
        wconv_kernel<<<dim3((unsigned)((n + 255) / 256)), dim3(256), 0, stream>>>(w, dst, K, N);
    };
    for (int l = 0; l < NDEP; ++l) {
        wconv(rg_in_w   + (size_t)l*Dm*Dm,    wT[l][0], Dm, Dm);
        wconv(rg_gate_w + (size_t)l*Dm*Dm,    wT[l][1], Dm, Dm);
        wconv(rg_out_w  + (size_t)l*Dm*Dm,    wT[l][2], Dm, Dm);
        wconv(qkv_w     + (size_t)l*Dm*3*Dm,  wT[l][3], Dm, 3*Dm);
        wconv(ao_w      + (size_t)l*Dm*Dm,    wT[l][4], Dm, Dm);
        wconv(mlp_w1    + (size_t)l*Dm*4*Dm,  wT[l][5], Dm, 4*Dm);
        wconv(mlp_w2    + (size_t)l*4*Dm*Dm,  wT[l][6], 4*Dm, Dm);
    }

    auto gemm = [&](int mode, const _Float16* A, const _Float16* Bt,
                    const float* bias, void* out, int N, int K) {
        dim3 grid((unsigned)(N / 256), (unsigned)(BT / 64));
        switch (mode) {
        case EPI_F32:    gemm_wmma<EPI_F32>   <<<grid, 256, 0, stream>>>(A, Bt, bias, out, BT, N, K); break;
        case EPI_SIG:    gemm_wmma<EPI_SIG>   <<<grid, 256, 0, stream>>>(A, Bt, bias, out, BT, N, K); break;
        case EPI_GELU16: gemm_wmma<EPI_GELU16><<<grid, 256, 0, stream>>>(A, Bt, bias, out, BT, N, K); break;
        default:         gemm_wmma<EPI_RES>   <<<grid, 256, 0, stream>>>(A, Bt, bias, out, BT, N, K); break;
        }
    };

    for (int l = 0; l < NDEP; ++l) {
        // ---- recurrent-gate block: x += rg_out(lru(in, gate)) ----
        ln_kernel<<<BT, 256, 0, stream>>>(xbuf, ln1_s + l*Dm, ln1_b + l*Dm, xln);
        gemm(EPI_F32, xln, wT[l][0], rg_in_b   + l*Dm, ubuf, Dm, Dm);
        gemm(EPI_SIG, xln, wT[l][1], rg_gate_b + l*Dm, gbuf, Dm, Dm);
        lru_scan<<<(Bb * Dm) / 256, 256, 0, stream>>>(gbuf, ubuf, xln); // h -> f16 (reuse xln)
        gemm(EPI_RES, xln, wT[l][2], rg_out_b + l*Dm, xbuf, Dm, Dm);

        // ---- local attention block: x += Wo * attn(qkv) ----
        ln_kernel<<<BT, 256, 0, stream>>>(xbuf, ln2_s + l*Dm, ln2_b + l*Dm, xln);
        gemm(EPI_F32, xln, wT[l][3], qkv_b + l*3*Dm, qkvb, 3*Dm, Dm);
        attn_kernel<<<(Bb * Hh * Tt * 32) / 256, 256, 0, stream>>>(qkvb, xln); // y -> f16
        gemm(EPI_RES, xln, wT[l][4], ao_b + l*Dm, xbuf, Dm, Dm);

        // ---- MLP block: x += W2 * gelu(W1 * ln3(x)) ----
        ln_kernel<<<BT, 256, 0, stream>>>(xbuf, ln3_s + l*Dm, ln3_b + l*Dm, xln);
        gemm(EPI_GELU16, xln, wT[l][5], mlp_b1 + l*4*Dm, hid, 4*Dm, Dm);
        gemm(EPI_RES,    hid, wT[l][6], mlp_b2 + l*Dm,   xbuf, Dm, 4*Dm);
    }

    (void)hipMemcpyAsync(d_out, xbuf, (size_t)BT * Dm * 4, hipMemcpyDeviceToDevice, stream);
}
